// LSHSelfAttention_59167469470175
// MI455X (gfx1250) — matrix operand order repaired and checked by reading, not compiled, not run on hardware
//
#include <hip/hip_runtime.h>
#include <hip/hip_bf16.h>

typedef __attribute__((ext_vector_type(16))) __bf16 v16bf;
typedef __attribute__((ext_vector_type(8)))  __bf16 bf16x8;
typedef __attribute__((ext_vector_type(8)))  float  v8f;

#define B_  2
#define N_  2048
#define D_  1024
#define H_  16
#define HD_ 64

union Frag {
  v16bf v;
  bf16x8 h[2];
};

static __device__ __forceinline__ v8f wmma_bf16(v16bf a, v16bf b, v8f c) {
  // D = A(16x32 bf16) * B(32x16 bf16) + C(f32)
  return __builtin_amdgcn_wmma_f32_16x16x32_bf16(false, a, false, b, (short)0, c,
                                                 false, false);
}

// Reductions across a 16-lane half (C-matrix columns live one-per-lane).
static __device__ __forceinline__ float redmax16(float v) {
  v = fmaxf(v, __shfl_xor(v, 1, 32));
  v = fmaxf(v, __shfl_xor(v, 2, 32));
  v = fmaxf(v, __shfl_xor(v, 4, 32));
  v = fmaxf(v, __shfl_xor(v, 8, 32));
  return v;
}
static __device__ __forceinline__ float redsum16(float v) {
  v += __shfl_xor(v, 1, 32);
  v += __shfl_xor(v, 2, 32);
  v += __shfl_xor(v, 4, 32);
  v += __shfl_xor(v, 8, 32);
  return v;
}
static __device__ __forceinline__ float redsum32(float v) {
  v = redsum16(v);
  v += __shfl_xor(v, 16, 32);
  return v;
}

__global__ void cvt_f32_bf16(const float* __restrict__ s, __bf16* __restrict__ d, int n) {
  for (int i = blockIdx.x * blockDim.x + threadIdx.x; i < n; i += gridDim.x * blockDim.x)
    d[i] = (__bf16)s[i];
}

// C[M,N] = A[M,K] @ B[N,K]^T (+ optional bias).  One wave -> 16x64 output.
// Both operands contract over their contiguous dim, so:
//   A-frag: lane holds row m0+l15, 16 contiguous K values at k0+hi*16
//   B-frag: lane holds row n0+c*16+l15, 16 contiguous K values at k0+hi*16
__global__ void gemm_bf16_wave(const __bf16* __restrict__ A, const __bf16* __restrict__ B,
                               float* __restrict__ C, const float* __restrict__ bias,
                               int M, int N, int K) {
  const int lane = threadIdx.x & 31;
  const int wid  = blockIdx.x * (blockDim.x >> 5) + (threadIdx.x >> 5);
  const int nT   = N >> 6;
  if (wid >= (M >> 4) * nT) return;
  const int m0  = (wid / nT) << 4;
  const int n0  = (wid % nT) << 6;
  const int l15 = lane & 15;
  const int hi  = lane >> 4;

  v8f acc[4] = {{0,0,0,0,0,0,0,0}, {0,0,0,0,0,0,0,0},
                {0,0,0,0,0,0,0,0}, {0,0,0,0,0,0,0,0}};

  const __bf16* ap  = A + (size_t)(m0 + l15) * K + hi * 16;
  const __bf16* bp0 = B + (size_t)(n0 +  0 + l15) * K + hi * 16;
  const __bf16* bp1 = B + (size_t)(n0 + 16 + l15) * K + hi * 16;
  const __bf16* bp2 = B + (size_t)(n0 + 32 + l15) * K + hi * 16;
  const __bf16* bp3 = B + (size_t)(n0 + 48 + l15) * K + hi * 16;

  for (int k0 = 0; k0 < K; k0 += 32) {
    Frag a;
    a.h[0] = *(const bf16x8*)(ap + k0);
    a.h[1] = *(const bf16x8*)(ap + k0 + 8);
    Frag b;
    b.h[0] = *(const bf16x8*)(bp0 + k0); b.h[1] = *(const bf16x8*)(bp0 + k0 + 8);
    acc[0] = wmma_bf16(a.v, b.v, acc[0]);
    b.h[0] = *(const bf16x8*)(bp1 + k0); b.h[1] = *(const bf16x8*)(bp1 + k0 + 8);
    acc[1] = wmma_bf16(a.v, b.v, acc[1]);
    b.h[0] = *(const bf16x8*)(bp2 + k0); b.h[1] = *(const bf16x8*)(bp2 + k0 + 8);
    acc[2] = wmma_bf16(a.v, b.v, acc[2]);
    b.h[0] = *(const bf16x8*)(bp3 + k0); b.h[1] = *(const bf16x8*)(bp3 + k0 + 8);
    acc[3] = wmma_bf16(a.v, b.v, acc[3]);
    __builtin_prefetch(ap + k0 + 128);   // global_prefetch_b8
  }

#pragma unroll
  for (int c = 0; c < 4; ++c) {
    const int col = n0 + c * 16 + l15;
    const float bv = bias ? bias[col] : 0.f;
#pragma unroll
    for (int r = 0; r < 8; ++r)
      C[(size_t)(m0 + r + hi * 8) * N + col] = acc[c][r] + bv;
  }
}

// qkv[b,n,3*D] f32 -> l2-normalized q,k as bf16 [b*h, n, hd] and v transposed
// to vt bf16 [b*h, hd, n].  One wave per (b, token, head): 64 elems = 32 lanes x2.
__global__ void norm_pack(const float* __restrict__ qkv, __bf16* __restrict__ qo,
                          __bf16* __restrict__ ko, __bf16* __restrict__ vto) {
  const int lane = threadIdx.x & 31;
  const int wid  = blockIdx.x * (blockDim.x >> 5) + (threadIdx.x >> 5);
  if (wid >= B_ * N_ * H_) return;
  const int h = wid & (H_ - 1);
  const int t = (wid >> 4) & (N_ - 1);
  const int b = wid >> 15;

  const float* base = qkv + ((size_t)b * N_ + t) * (3 * D_) + h * HD_ + lane * 2;
  float2 qv = *(const float2*)(base);
  float2 kv = *(const float2*)(base + D_);
  float2 vv = *(const float2*)(base + 2 * D_);

  float qn = redsum32(qv.x * qv.x + qv.y * qv.y);
  float kn = redsum32(kv.x * kv.x + kv.y * kv.y);
  float qi = 1.f / fmaxf(sqrtf(qn), 1e-12f);
  float ki = 1.f / fmaxf(sqrtf(kn), 1e-12f);

  const int bh = b * H_ + h;
  __bf16* qr = qo + ((size_t)bh * N_ + t) * HD_ + lane * 2;
  qr[0] = (__bf16)(qv.x * qi); qr[1] = (__bf16)(qv.y * qi);
  __bf16* kr = ko + ((size_t)bh * N_ + t) * HD_ + lane * 2;
  kr[0] = (__bf16)(kv.x * ki); kr[1] = (__bf16)(kv.y * ki);
  vto[((size_t)bh * HD_ + lane * 2    ) * N_ + t] = (__bf16)vv.x;
  vto[((size_t)bh * HD_ + lane * 2 + 1) * N_ + t] = (__bf16)vv.y;
}

// Flash attention. One wave owns a 16-row q tile; streams keys in chunks of 32.
// S tile = Q(16x64) @ K^T -> 2 f32 accumulators (2 WMMA each).  Online softmax
// state (row max / row sum) lives per 16-lane half.  P goes through LDS to
// convert C-layout -> A-layout, then O += P(16x32) @ Vt-chunk (4 WMMA).
__global__ void attn_fwd(const __bf16* __restrict__ q, const __bf16* __restrict__ k,
                         const __bf16* __restrict__ vt, const int* __restrict__ mask,
                         __bf16* __restrict__ out) {
  __shared__ __bf16 pbuf[8][16 * 32];
  const int lane = threadIdx.x & 31;
  const int w    = threadIdx.x >> 5;
  const int wid  = blockIdx.x * (blockDim.x >> 5) + w;
  if (wid >= B_ * H_ * (N_ / 16)) return;
  const int qt = wid & 127;        // N_/16 tiles
  const int bh = wid >> 7;
  const int b  = bh >> 4;
  const int h  = bh & 15;
  const int q0 = qt << 4;
  const int l15 = lane & 15;
  const int hi  = lane >> 4;

  // Q fragments for d 0..31 and 32..63, reused for every key chunk.
  const __bf16* qbase = q + ((size_t)bh * N_ + q0 + l15) * HD_ + hi * 16;
  Frag a0, a1;
  a0.h[0] = *(const bf16x8*)(qbase);      a0.h[1] = *(const bf16x8*)(qbase + 8);
  a1.h[0] = *(const bf16x8*)(qbase + 32); a1.h[1] = *(const bf16x8*)(qbase + 40);

  v8f o[4] = {{0,0,0,0,0,0,0,0}, {0,0,0,0,0,0,0,0},
              {0,0,0,0,0,0,0,0}, {0,0,0,0,0,0,0,0}};
  float mrow[8], lrow[8];
#pragma unroll
  for (int r = 0; r < 8; ++r) { mrow[r] = -3.0e38f; lrow[r] = 0.f; }

  const __bf16* kb   = k  + (size_t)bh * N_ * HD_;
  const __bf16* vtb  = vt + (size_t)bh * HD_ * N_;
  const int*    mptr = mask + b * N_;
  __bf16* pw = &pbuf[w][0];

  for (int j0 = 0; j0 < N_; j0 += 32) {
    const __bf16* kp0 = kb + (size_t)(j0 +      l15) * HD_ + hi * 16;
    const __bf16* kp1 = kb + (size_t)(j0 + 16 + l15) * HD_ + hi * 16;
    v8f s0 = {0,0,0,0,0,0,0,0}, s1 = {0,0,0,0,0,0,0,0};
    Frag bk;
    bk.h[0] = *(const bf16x8*)(kp0);      bk.h[1] = *(const bf16x8*)(kp0 + 8);
    s0 = wmma_bf16(a0.v, bk.v, s0);
    bk.h[0] = *(const bf16x8*)(kp0 + 32); bk.h[1] = *(const bf16x8*)(kp0 + 40);
    s0 = wmma_bf16(a1.v, bk.v, s0);
    bk.h[0] = *(const bf16x8*)(kp1);      bk.h[1] = *(const bf16x8*)(kp1 + 8);
    s1 = wmma_bf16(a0.v, bk.v, s1);
    bk.h[0] = *(const bf16x8*)(kp1 + 32); bk.h[1] = *(const bf16x8*)(kp1 + 40);
    s1 = wmma_bf16(a1.v, bk.v, s1);

    const float mf0 = mptr[j0 + l15]      ? 1.f : 0.f;
    const float mf1 = mptr[j0 + 16 + l15] ? 1.f : 0.f;
#pragma unroll
    for (int r = 0; r < 8; ++r) {
      float x0 = s0[r] * 0.125f; if (mf0 == 0.f) x0 = -3.0e38f;
      float x1 = s1[r] * 0.125f; if (mf1 == 0.f) x1 = -3.0e38f;
      float rm = redmax16(fmaxf(x0, x1));
      float nm = fmaxf(mrow[r], rm);
      float alpha = __expf(mrow[r] - nm);
      mrow[r] = nm;
      float p0 = __expf(x0 - nm) * mf0;
      float p1 = __expf(x1 - nm) * mf1;
      lrow[r] = lrow[r] * alpha + redsum16(p0 + p1);
      o[0][r] *= alpha; o[1][r] *= alpha; o[2][r] *= alpha; o[3][r] *= alpha;
      const int row = r + hi * 8;
      pw[row * 32 + l15]      = (__bf16)p0;
      pw[row * 32 + 16 + l15] = (__bf16)p1;
    }
    // LDS ops are in-order within a wave; make the transpose RAW explicit.
    asm volatile("s_wait_dscnt 0" ::: "memory");

    Frag pf;  // A-fragment of P(16x32): row l15, cols {hi*16..hi*16+15}
    pf.h[0] = *(const bf16x8*)(pw + l15 * 32 + hi * 16);
    pf.h[1] = *(const bf16x8*)(pw + l15 * 32 + hi * 16 + 8);
#pragma unroll
    for (int c = 0; c < 4; ++c) {
      const __bf16* vp = vtb + (size_t)(c * 16 + l15) * N_ + j0 + hi * 16;
      Frag bv;
      bv.h[0] = *(const bf16x8*)vp; bv.h[1] = *(const bf16x8*)(vp + 8);
      o[c] = wmma_bf16(pf.v, bv.v, o[c]);
    }
  }

#pragma unroll
  for (int r = 0; r < 8; ++r) {
    const float inv = lrow[r] > 0.f ? 1.f / lrow[r] : 0.f;
    const size_t base = ((size_t)b * N_ + q0 + r + hi * 8) * D_ + h * HD_ + l15;
    out[base +  0] = (__bf16)(o[0][r] * inv);
    out[base + 16] = (__bf16)(o[1][r] * inv);
    out[base + 32] = (__bf16)(o[2][r] * inv);
    out[base + 48] = (__bf16)(o[3][r] * inv);
  }
}

extern "C" void kernel_launch(void* const* d_in, const int* in_sizes, int n_in,
                              void* d_out, int out_size, void* d_ws, size_t ws_size,
                              hipStream_t stream) {
  const float* x     = (const float*)d_in[0];
  const int*   msk   = (const int*)d_in[1];
  const float* w_qkv = (const float*)d_in[2];
  const float* w_out = (const float*)d_in[3];
  const float* b_out = (const float*)d_in[4];
  float* out = (float*)d_out;

  char* ws = (char*)d_ws;
  __bf16* xb   = (__bf16*)(ws + 0);          //  8 MB  x bf16 [4096,1024]
  __bf16* wqb  = (__bf16*)(ws + 8388608);    //  6 MB  w_qkv bf16 [3072,1024]
  __bf16* wob  = (__bf16*)(ws + 14680064);   //  2 MB  w_out bf16 [1024,1024]
  float*  qkvf = (float*)(ws + 16777216);    // 48 MB  qkv f32 [4096,3072]
  __bf16* qb   = (__bf16*)(ws + 67108864);   //  8 MB  q bf16 [32,2048,64]
  __bf16* kbuf = (__bf16*)(ws + 75497472);   //  8 MB  k bf16 [32,2048,64]
  __bf16* vtb  = (__bf16*)(ws + 83886080);   //  8 MB  v^T bf16 [32,64,2048]
  __bf16* ao   = (__bf16*)(ws + 92274688);   //  8 MB  attn out bf16 [4096,1024]

  cvt_f32_bf16<<<2048, 256, 0, stream>>>(x, xb, B_ * N_ * D_);
  cvt_f32_bf16<<<2048, 256, 0, stream>>>(w_qkv, wqb, 3 * D_ * D_);
  cvt_f32_bf16<<<1024, 256, 0, stream>>>(w_out, wob, D_ * D_);

  // qkv = x @ w_qkv^T : waves = (4096/16)*(3072/64) = 12288 -> 1536 blocks
  gemm_bf16_wave<<<1536, 256, 0, stream>>>(xb, wqb, qkvf, nullptr,
                                           B_ * N_, 3 * D_, D_);
  // l2norm + head repack + V transpose: 65536 waves -> 8192 blocks
  norm_pack<<<8192, 256, 0, stream>>>(qkvf, qb, kbuf, vtb);
  // flash attention: 2*16*128 = 4096 waves -> 512 blocks
  attn_fwd<<<512, 256, 0, stream>>>(qb, kbuf, vtb, msk, ao);
  // out = attn @ w_out^T + b_out : 256*16 = 4096 waves -> 512 blocks
  gemm_bf16_wave<<<512, 256, 0, stream>>>(ao, wob, out, b_out,
                                          B_ * N_, D_, D_);
}
